// UniversalReasoner_52845277609988
// MI455X (gfx1250) — matrix operand order repaired and checked by reading, not compiled, not run on hardware
//
#include <hip/hip_runtime.h>
#include <hip/hip_bf16.h>

// ---------------- model constants ----------------
#define BB   16
#define SS   128
#define SCRN 64
#define TT   (SS + SCRN)     // 192
#define DD   1024
#define HH   8
#define HD   128
#define NSTEPS 8
#define VV   100277
#define DFF  4096
#define NEGV (-1000000000.0f)

typedef __bf16 v16bf __attribute__((ext_vector_type(16)));
typedef __bf16 v2bf  __attribute__((ext_vector_type(2)));
typedef float  v8f   __attribute__((ext_vector_type(8)));

#if defined(__has_builtin)
#if __has_builtin(__builtin_amdgcn_cvt_pk_bf16_f32)
#define HAVE_CVT_PK_BF16 1
#endif
#endif

// pack two f32 -> two bf16 (RNE) in one dword, pure 32-bit ALU ops
__device__ __forceinline__ unsigned int pk2bf(float lo, float hi) {
#ifdef HAVE_CVT_PK_BF16
    union { v2bf v; unsigned int u; } cv;
    cv.v = __builtin_amdgcn_cvt_pk_bf16_f32(lo, hi);
    return cv.u;
#else
    unsigned int ul = __float_as_uint(lo);
    unsigned int uh = __float_as_uint(hi);
    ul += 0x7FFFu + ((ul >> 16) & 1u);
    uh += 0x7FFFu + ((uh >> 16) & 1u);
    return (ul >> 16) | (uh & 0xFFFF0000u);
#endif
}

// gelu(v) = v * sigmoid(1.5957691*(v + 0.044715 v^3))  (== tanh-approx gelu)
__device__ __forceinline__ float gelu_fast(float v) {
    float z = 1.5957691216057308f * (v + 0.044715f * v * v * v);
    return v / (1.f + __expf(-z));
}

// ---------------- WMMA bf16 GEMM ----------------
// C = act(A(MxK,f32) @ B(KxN,f32) [+bias]) [+residual], f32 out.
// 256 threads = 8 waves in a 4(M) x 2(N) grid; block tile 64(M) x 128(N);
// grid is (M-tiles, N-tiles) so M varies fastest -> B slices stay hot in L2.
// LDS double-buffered K-chunks of 32; 4 v_wmma per wave per chunk, A reused.
// B staging: each thread owns a 4col x 4k sub-tile -> packed ds_store_b64,
// bank-conflict-free ((16*colgrp + 2*kslot) mod 64 all distinct per wave).
#define BM 64
#define BN 128
#define BK 32

__global__ __launch_bounds__(256)
void gemm_bf16_wmma(const float* __restrict__ A, const float* __restrict__ Bm,
                    const float* __restrict__ bias, const float* __restrict__ residual,
                    float* __restrict__ C, int M, int N, int K, int apply_gelu)
{
    __shared__ unsigned short As[2][BM][BK + 8];   // [buf][m][k]
    __shared__ unsigned short Bs[2][BN][BK + 8];   // [buf][n][k] (transposed)

    const int tid  = threadIdx.x;
    const int lane = tid & 31;
    const int wave = tid >> 5;
    const int wm   = wave >> 1;        // 0..3
    const int wn   = wave & 1;         // 0..1
    const int m0   = blockIdx.x * BM;  // M fastest-varying across blocks
    const int n0   = blockIdx.y * BN;
    const bool interior = (n0 + BN) <= N;

    union FragAB { v16bf v; unsigned int u[8]; };
    union FragC  { v8f v;  float f[8]; };
    FragC acc[4];
#pragma unroll
    for (int t = 0; t < 4; ++t)
#pragma unroll
        for (int j = 0; j < 8; ++j) acc[t].f[j] = 0.f;

    const int arow  = (lane & 15) + 16 * wm;
    const int abase = (lane & 16) ? 8 : 0;     // ISA 16-bit A 16x32 layout
    const int bbase = (lane & 16) ? 8 : 0;     // ISA 16-bit B 32x16 layout (same k order as A)

    // staging coordinates
    const int a_e0 = tid * 8;            // 2048 A elems / 256 thr
    const int a_r  = a_e0 >> 5;
    const int a_k0 = a_e0 & 31;
    const int b_kk = (tid & 7) * 4;      // k-slot: 4 consecutive k
    const int b_c0 = (tid >> 3) * 4;     // col-group: 4 consecutive cols

    // ---- prologue: stage chunk 0 into buffer 0 ----
    {
        const float* ap = &A[(long)(m0 + a_r) * K + a_k0];
        float4 v0 = *(const float4*)(ap);
        float4 v1 = *(const float4*)(ap + 4);
        unsigned int* asd = (unsigned int*)&As[0][a_r][a_k0];
        asd[0] = pk2bf(v0.x, v0.y); asd[1] = pk2bf(v0.z, v0.w);
        asd[2] = pk2bf(v1.x, v1.y); asd[3] = pk2bf(v1.z, v1.w);

        float bv[4][4];
        if (interior) {
#pragma unroll
            for (int c = 0; c < 4; ++c)
#pragma unroll
                for (int i = 0; i < 4; ++i)
                    bv[c][i] = Bm[(long)(b_kk + i) * N + (n0 + b_c0 + c)];
        } else {
#pragma unroll
            for (int c = 0; c < 4; ++c)
#pragma unroll
                for (int i = 0; i < 4; ++i)
                    bv[c][i] = ((n0 + b_c0 + c) < N)
                             ? Bm[(long)(b_kk + i) * N + (n0 + b_c0 + c)] : 0.f;
        }
#pragma unroll
        for (int c = 0; c < 4; ++c) {
            uint2 pk;
            pk.x = pk2bf(bv[c][0], bv[c][1]);
            pk.y = pk2bf(bv[c][2], bv[c][3]);
            *(uint2*)&Bs[0][b_c0 + c][b_kk] = pk;
        }
    }
    __syncthreads();

    const int nk = K / BK;
    for (int kc = 0; kc < nk; ++kc) {
        const int cur = kc & 1, nxt = cur ^ 1;
        const int k_next = (kc + 1) * BK;
        const bool more = k_next < K;

        // ---- issue next chunk's global loads early (overlap with WMMA) ----
        float4 av0, av1;
        float bv[4][4];
        if (more) {
            const float* ap = &A[(long)(m0 + a_r) * K + k_next + a_k0];
            av0 = *(const float4*)(ap);
            av1 = *(const float4*)(ap + 4);
            if (interior) {
#pragma unroll
                for (int c = 0; c < 4; ++c)
#pragma unroll
                    for (int i = 0; i < 4; ++i)
                        bv[c][i] = Bm[(long)(k_next + b_kk + i) * N + (n0 + b_c0 + c)];
            } else {
#pragma unroll
                for (int c = 0; c < 4; ++c)
#pragma unroll
                    for (int i = 0; i < 4; ++i)
                        bv[c][i] = ((n0 + b_c0 + c) < N)
                                 ? Bm[(long)(k_next + b_kk + i) * N + (n0 + b_c0 + c)] : 0.f;
            }
        }

        // ---- gather fragments from current buffer ----
        FragAB fa;
#pragma unroll
        for (int j = 0; j < 4; ++j)
            fa.u[j]     = *(const unsigned int*)&As[cur][arow][abase + 2 * j];
#pragma unroll
        for (int j = 0; j < 4; ++j)
            fa.u[4 + j] = *(const unsigned int*)&As[cur][arow][16 + abase + 2 * j];

        FragAB fb[4];
#pragma unroll
        for (int t = 0; t < 4; ++t) {
            const int bcol = 64 * wn + 16 * t + (lane & 15);
#pragma unroll
            for (int j = 0; j < 8; ++j)
                fb[t].u[j] = *(const unsigned int*)&Bs[cur][bcol][bbase + 2 * j + ((j >= 4) ? 8 : 0)];
        }
#pragma unroll
        for (int t = 0; t < 4; ++t)
            acc[t].v = __builtin_amdgcn_wmma_f32_16x16x32_bf16(
                           false, fa.v, false, fb[t].v, (short)0, acc[t].v, false, false);

        // ---- convert + store next chunk into alternate buffer ----
        if (more) {
            unsigned int* asd = (unsigned int*)&As[nxt][a_r][a_k0];
            asd[0] = pk2bf(av0.x, av0.y); asd[1] = pk2bf(av0.z, av0.w);
            asd[2] = pk2bf(av1.x, av1.y); asd[3] = pk2bf(av1.z, av1.w);
#pragma unroll
            for (int c = 0; c < 4; ++c) {
                uint2 pk;
                pk.x = pk2bf(bv[c][0], bv[c][1]);
                pk.y = pk2bf(bv[c][2], bv[c][3]);
                *(uint2*)&Bs[nxt][b_c0 + c][b_kk] = pk;
            }
        }
        __syncthreads();
    }

    // ---- epilogue ----
    const int rbase = m0 + 16 * wm + ((lane & 16) ? 8 : 0);
#pragma unroll
    for (int t = 0; t < 4; ++t) {
        const int ccol = n0 + 64 * wn + 16 * t + (lane & 15);
        if (ccol < N) {
            float bval = bias ? bias[ccol] : 0.f;
#pragma unroll
            for (int j = 0; j < 8; ++j) {
                long idx = (long)(rbase + j) * N + ccol;
                float val = acc[t].f[j] + bval;
                if (apply_gelu) val = gelu_fast(val);
                if (residual) val += residual[idx];
                C[idx] = val;
            }
        }
    }
}

// ---------------- embedding gather + scratch tile ----------------
__global__ void embed_kernel(const int* __restrict__ tokens, const float* __restrict__ embed,
                             const float* __restrict__ scratch, float* __restrict__ x)
{
    long idx = (long)blockIdx.x * blockDim.x + threadIdx.x;
    long total = (long)BB * TT * DD;
    if (idx >= total) return;
    int d  = (int)(idx % DD);
    long bt = idx / DD;
    int t  = (int)(bt % TT);
    int b  = (int)(bt / TT);
    float v;
    if (t < SS) { int tok = tokens[b * SS + t]; v = embed[(long)tok * DD + d]; }
    else        { v = scratch[(long)(t - SS) * DD + d]; }
    x[idx] = v;
}

__global__ void add_time_kernel(float* __restrict__ x, const float* __restrict__ te, int step)
{
    long idx = (long)blockIdx.x * blockDim.x + threadIdx.x;
    long total = (long)BB * TT * DD;
    if (idx >= total) return;
    x[idx] += te[(long)step * DD + (idx % DD)];
}

// ---------------- layernorm (one token row per block) ----------------
__global__ __launch_bounds__(256)
void ln_kernel(const float* __restrict__ x, const float* __restrict__ s,
               const float* __restrict__ b, float* __restrict__ out)
{
    __shared__ float red[256];
    const int row = blockIdx.x, tid = threadIdx.x;
    const float* xr = x + (long)row * DD;
    float p = 0.f;
#pragma unroll
    for (int i = tid; i < DD; i += 256) p += xr[i];
    red[tid] = p; __syncthreads();
    for (int o = 128; o > 0; o >>= 1) { if (tid < o) red[tid] += red[tid + o]; __syncthreads(); }
    float mu = red[0] / DD; __syncthreads();
    p = 0.f;
#pragma unroll
    for (int i = tid; i < DD; i += 256) { float dlt = xr[i] - mu; p += dlt * dlt; }
    red[tid] = p; __syncthreads();
    for (int o = 128; o > 0; o >>= 1) { if (tid < o) red[tid] += red[tid + o]; __syncthreads(); }
    float rsig = rsqrtf(red[0] / DD + 1e-6f);
    for (int i = tid; i < DD; i += 256)
        out[(long)row * DD + i] = (xr[i] - mu) * rsig * s[i] + b[i];
}

// ---------------- RoPE on q and k ----------------
__global__ void rope_kernel(float* __restrict__ q, float* __restrict__ k)
{
    long idx = (long)blockIdx.x * blockDim.x + threadIdx.x;
    long total = (long)BB * TT * HH * (HD / 2);
    if (idx >= total) return;
    int p   = (int)(idx % (HD / 2));
    long r  = idx / (HD / 2);
    int h   = (int)(r % HH);
    long bt = r / HH;
    int t   = (int)(bt % TT);
    float inv = __expf(-((2.f * p) / (float)HD) * 9.210340371976184f); // ln(10000)
    float f = (float)t * inv, c, sn;
    __sincosf(f, &sn, &c);
    long base = bt * DD + (long)h * HD + 2 * p;
    float qe = q[base], qo = q[base + 1];
    q[base]     = qe * c - qo * sn;
    q[base + 1] = qo * c + qe * sn;
    float ke = k[base], ko = k[base + 1];
    k[base]     = ke * c - ko * sn;
    k[base + 1] = ko * c + ke * sn;
}

// ---------------- attention logits (masked) ----------------
__global__ void attn_logits_kernel(const float* __restrict__ q, const float* __restrict__ k,
                                   float* __restrict__ logits)
{
    long idx = (long)blockIdx.x * blockDim.x + threadIdx.x;
    long total = (long)BB * HH * TT * TT;
    if (idx >= total) return;
    int t = (int)(idx % TT);
    int s = (int)((idx / TT) % TT);
    int h = (int)((idx / ((long)TT * TT)) % HH);
    int b = (int)(idx / ((long)TT * TT * HH));
    bool keep = (t <= s) || (s >= SS);
    float out = NEGV;
    if (keep) {
        const float* qp = q + ((long)b * TT + s) * DD + (long)h * HD;
        const float* kp = k + ((long)b * TT + t) * DD + (long)h * HD;
        float acc = 0.f;
#pragma unroll 8
        for (int d = 0; d < HD; ++d) acc += qp[d] * kp[d];
        out = acc * 0.08838834764831845f;   // 1/sqrt(128)
    }
    logits[idx] = out;
}

// ---------------- softmax over last dim (row of T per block) ----------------
__global__ __launch_bounds__(256)
void softmax_kernel(float* __restrict__ logits)
{
    __shared__ float red[256];
    float* row = logits + (long)blockIdx.x * TT;
    int tid = threadIdx.x;
    float v = (tid < TT) ? row[tid] : NEGV;
    red[tid] = v; __syncthreads();
    for (int o = 128; o > 0; o >>= 1) { if (tid < o) red[tid] = fmaxf(red[tid], red[tid + o]); __syncthreads(); }
    float mx = red[0]; __syncthreads();
    float e = (tid < TT) ? __expf(v - mx) : 0.f;
    red[tid] = e; __syncthreads();
    for (int o = 128; o > 0; o >>= 1) { if (tid < o) red[tid] += red[tid + o]; __syncthreads(); }
    float inv = 1.f / red[0];
    if (tid < TT) row[tid] = e * inv;
}

// ---------------- o = softmax(w) @ v ----------------
__global__ void attn_av_kernel(const float* __restrict__ w, const float* __restrict__ v,
                               float* __restrict__ o)
{
    long idx = (long)blockIdx.x * blockDim.x + threadIdx.x;
    long total = (long)BB * TT * DD;
    if (idx >= total) return;
    int d  = (int)(idx % HD);
    long r = idx / HD;
    int h  = (int)(r % HH);
    long bs = r / HH;
    int s  = (int)(bs % TT);
    int b  = (int)(bs / TT);
    const float* wp = w + (((long)b * HH + h) * TT + s) * TT;
    const float* vp = v + (long)b * TT * DD + (long)h * HD + d;
    float acc = 0.f;
#pragma unroll 4
    for (int t = 0; t < TT; ++t) acc += wp[t] * vp[(long)t * DD];
    o[((long)b * TT + s) * DD + (long)h * HD + d] = acc;
}

// ---------------- halting signal per batch ----------------
__global__ __launch_bounds__(256)
void halt_kernel(const float* __restrict__ x, const float* __restrict__ hw,
                 const float* __restrict__ hb, float* __restrict__ out, int step)
{
    __shared__ float red[256];
    int b = blockIdx.x, tid = threadIdx.x;
    float p = 0.f;
    for (int e = tid; e < SCRN * DD; e += 256) {
        int t = e / DD, d = e % DD;
        p += x[((long)b * TT + SS + t) * DD + d] * hw[d];
    }
    red[tid] = p; __syncthreads();
    for (int o = 128; o > 0; o >>= 1) { if (tid < o) red[tid] += red[tid + o]; __syncthreads(); }
    if (tid == 0) {
        float v = red[0] / (float)SCRN + hb[0];
        out[(long)step * BB + b] = 1.f / (1.f + __expf(-v));
    }
}

// ---------------- pack decoder rows x[:, :S, :] ----------------
__global__ void xdec_kernel(const float* __restrict__ x, float* __restrict__ xd)
{
    long idx = (long)blockIdx.x * blockDim.x + threadIdx.x;
    long total = (long)BB * SS * DD;
    if (idx >= total) return;
    int d  = (int)(idx % DD);
    long bs = idx / DD;
    int s  = (int)(bs % SS);
    int b  = (int)(bs / SS);
    xd[idx] = x[((long)b * TT + s) * DD + d];
}

// ---------------- launch ----------------
extern "C" void kernel_launch(void* const* d_in, const int* in_sizes, int n_in,
                              void* d_out, int out_size, void* d_ws, size_t ws_size,
                              hipStream_t stream)
{
    const int*   tokens = (const int*)  d_in[0];
    const float* embed  = (const float*)d_in[1];
    const float* temb   = (const float*)d_in[2];
    const float* scr    = (const float*)d_in[3];
    const float* wq = (const float*)d_in[4];  const float* bq = (const float*)d_in[5];
    const float* wk = (const float*)d_in[6];  const float* bk = (const float*)d_in[7];
    const float* wv = (const float*)d_in[8];  const float* bv = (const float*)d_in[9];
    const float* wo = (const float*)d_in[10]; const float* bo = (const float*)d_in[11];
    const float* l1s = (const float*)d_in[12]; const float* l1b = (const float*)d_in[13];
    const float* l2s = (const float*)d_in[14]; const float* l2b = (const float*)d_in[15];
    const float* w1 = (const float*)d_in[16]; const float* b1 = (const float*)d_in[17];
    const float* w2 = (const float*)d_in[18]; const float* b2 = (const float*)d_in[19];
    const float* hw = (const float*)d_in[20]; const float* hb = (const float*)d_in[21];
    const float* dw = (const float*)d_in[22]; const float* db = (const float*)d_in[23];

    float* outp = (float*)d_out;
    const long BTD = (long)BB * TT * DD;           // 3,145,728
    float* x   = (float*)d_ws;
    float* h   = x   + BTD;
    float* q   = h   + BTD;
    float* k   = q   + BTD;
    float* v   = k   + BTD;
    float* o   = v   + BTD;
    float* lg  = o   + BTD;                        // B*H*T*T
    float* mid = lg  + (long)BB * HH * TT * TT;    // B*T*DFF
    float* xd  = mid + (long)BB * TT * DFF;        // B*S*D

    const long LOGITS = (long)BB * SS * VV;
    float* halts = outp + LOGITS;

    const int TPB = 256;
    dim3 blk(TPB);
    long nBTD = BTD;

    embed_kernel<<<dim3((nBTD + TPB - 1) / TPB), blk, 0, stream>>>(tokens, embed, scr, x);

    // grid = (M-tiles, N-tiles): M varies fastest -> each B slice reused from L2
    dim3 gD ((BB * TT) / BM, DD  / BN);   // (48, 8)
    dim3 gFF((BB * TT) / BM, DFF / BN);   // (48, 32)

    for (int step = 0; step < NSTEPS; ++step) {
        add_time_kernel<<<dim3((nBTD + TPB - 1) / TPB), blk, 0, stream>>>(x, temb, step);
        ln_kernel<<<dim3(BB * TT), blk, 0, stream>>>(x, l1s, l1b, h);

        gemm_bf16_wmma<<<gD, blk, 0, stream>>>(h, wq, bq, nullptr, q, BB * TT, DD, DD, 0);
        gemm_bf16_wmma<<<gD, blk, 0, stream>>>(h, wk, bk, nullptr, k, BB * TT, DD, DD, 0);
        gemm_bf16_wmma<<<gD, blk, 0, stream>>>(h, wv, bv, nullptr, v, BB * TT, DD, DD, 0);

        long nRope = (long)BB * TT * HH * (HD / 2);
        rope_kernel<<<dim3((nRope + TPB - 1) / TPB), blk, 0, stream>>>(q, k);

        long nLog = (long)BB * HH * TT * TT;
        attn_logits_kernel<<<dim3((nLog + TPB - 1) / TPB), blk, 0, stream>>>(q, k, lg);
        softmax_kernel<<<dim3(BB * HH * TT), blk, 0, stream>>>(lg);
        attn_av_kernel<<<dim3((nBTD + TPB - 1) / TPB), blk, 0, stream>>>(lg, v, o);

        gemm_bf16_wmma<<<gD, blk, 0, stream>>>(o, wo, bo, x, x, BB * TT, DD, DD, 0);

        ln_kernel<<<dim3(BB * TT), blk, 0, stream>>>(x, l2s, l2b, h);
        // GELU fused into epilogue of the w1 GEMM
        gemm_bf16_wmma<<<gFF, blk, 0, stream>>>(h, w1, b1, nullptr, mid, BB * TT, DFF, DD, 1);
        gemm_bf16_wmma<<<gD, blk, 0, stream>>>(mid, w2, b2, x, x, BB * TT, DD, DFF, 0);

        halt_kernel<<<dim3(BB), blk, 0, stream>>>(x, hw, hb, halts, step);
    }

    long nXd = (long)BB * SS * DD;
    xdec_kernel<<<dim3((nXd + TPB - 1) / TPB), blk, 0, stream>>>(x, xd);
    dim3 gDec((BB * SS) / BM, (VV + BN - 1) / BN);   // (32, 784)
    gemm_bf16_wmma<<<gDec, blk, 0, stream>>>(xd, dw, db, nullptr, outp, BB * SS, VV, DD, 0);
}
